// GNNTransformerModel_30820685316134
// MI455X (gfx1250) — compile-verified, matrix operand order checked by reading, not run on hardware
//
#include <hip/hip_runtime.h>
#include <hip/hip_bf16.h>
#include <math.h>

typedef __attribute__((ext_vector_type(16))) __bf16 v16bf;
typedef __attribute__((ext_vector_type(8)))  __bf16 v8bf;
typedef __attribute__((ext_vector_type(4)))  __bf16 v4bf;
typedef __attribute__((ext_vector_type(8)))  float  v8f;
typedef __attribute__((ext_vector_type(4)))  float  v4f;

#define NN 200000   // nodes
#define NE 400000   // edges
#define NG 8000     // graphs

#define ASTRIDE 40  // LDS row stride in bf16 elems (80B: 16B-aligned, bank-spread)

// =====================================================================
// WMMA GEMM: C[M,N] = (A[M,K] @ W[K,N]) + bias, optional relu.
// f32 in memory, bf16 in LDS, f32 accumulate via v_wmma_f32_16x16x32_bf16.
// Block = 128 threads = 4 waves. Block tile 64(M) x 64(N); each wave
// computes 16 rows x 64 cols = 4 wmma accumulators. Requires K % 4 == 0
// (true for all layers: 16/32/64/512/1024).
// A-tile rows and B-tile *columns* stored contiguously in LDS so each
// fragment is two ds_load_b128 per lane (16-bit A 16x32 layout:
// lane half hf -> K base hf*8, elements {0..7} then {16..23}).
// =====================================================================
__global__ void gemm_bf16_wmma(const float* __restrict__ A,
                               const float* __restrict__ W,
                               const float* __restrict__ bias,
                               float* __restrict__ C,
                               int M, int K, int N, int relu)
{
    __shared__ __bf16 As[64 * ASTRIDE];   // [row][k]     row-major
    __shared__ __bf16 Bt[64 * ASTRIDE];   // [col][k]     transposed

    const int tid  = threadIdx.x;
    const int wv   = tid >> 5;
    const int lane = tid & 31;
    const int hf   = lane >> 4;
    const int rl   = lane & 15;
    const int rowBase = blockIdx.x * 64;
    const int colBase = blockIdx.y * 64;

    v8f acc[4] = {};

    for (int k0 = 0; k0 < K; k0 += 32) {
        // ---- stage A: 64 rows x 32 k. 2 threads/row, 4x float4 each ----
        {
            int r  = tid >> 1;
            int cb = (tid & 1) * 16;
            int gr = rowBase + r;
            const float* ap = A + (size_t)min(gr, M - 1) * K;
            #pragma unroll
            for (int j = 0; j < 4; ++j) {
                int c  = cb + j * 4;
                int gk = k0 + c;
                bool ok = (gr < M) && (gk < K);
                v4f va = *(const v4f*)(ap + (ok ? gk : 0));
                v4bf pb;
                #pragma unroll
                for (int u = 0; u < 4; ++u)
                    pb[u] = (__bf16)(ok ? va[u] : 0.0f);
                *(v4bf*)&As[r * ASTRIDE + c] = pb;
            }
        }
        // ---- stage B transposed: 32 k x 64 n -> Bt[col][k] ----
        {
            int r  = tid >> 2;             // k row within tile
            int nb = (tid & 3) * 16;       // col offset within tile
            int gk = k0 + r;
            const float* wp = W + (size_t)min(gk, K - 1) * N;
            #pragma unroll
            for (int j = 0; j < 16; ++j) {
                int c  = nb + j;
                int gc = colBase + c;
                bool ok = (gk < K) && (gc < N);
                float v = wp[ok ? gc : 0];
                Bt[c * ASTRIDE + r] = (__bf16)(ok ? v : 0.0f);
            }
        }
        __syncthreads();

        // ---- fragments: two b128 LDS loads each ----
        const __bf16* ar = &As[(wv * 16 + rl) * ASTRIDE + hf * 8];
        v8bf alo = *(const v8bf*)(ar);
        v8bf ahi = *(const v8bf*)(ar + 16);
        v16bf af = __builtin_shufflevector(alo, ahi,
                    0,1,2,3,4,5,6,7,8,9,10,11,12,13,14,15);
        #pragma unroll
        for (int nt = 0; nt < 4; ++nt) {
            const __bf16* br = &Bt[(nt * 16 + rl) * ASTRIDE + hf * 8];
            v8bf blo = *(const v8bf*)(br);
            v8bf bhi = *(const v8bf*)(br + 16);
            v16bf bfv = __builtin_shufflevector(blo, bhi,
                        0,1,2,3,4,5,6,7,8,9,10,11,12,13,14,15);
            acc[nt] = __builtin_amdgcn_wmma_f32_16x16x32_bf16(
                          false, af, false, bfv, (short)0, acc[nt], false, false);
        }
        __syncthreads();
    }

    // ---- store: C/D layout lane->col, VGPR r -> M = r + 8*hf ----
    #pragma unroll
    for (int nt = 0; nt < 4; ++nt) {
        int gc = colBase + nt * 16 + rl;
        #pragma unroll
        for (int r = 0; r < 8; ++r) {
            int gr = rowBase + wv * 16 + hf * 8 + r;
            if (gr < M && gc < N) {
                float v = acc[nt][r];
                if (bias) v += bias[gc];
                if (relu) v = fmaxf(v, 0.0f);
                C[(size_t)gr * N + gc] = v;
            }
        }
    }
}

// ===================== graph / attention kernels =====================
__device__ __forceinline__ int f32_order(float f) {
    int i = __float_as_int(f);
    return (i >= 0) ? i : (i ^ 0x7FFFFFFF);   // monotone int encoding
}
__device__ __forceinline__ float order_f32(int k) {
    int i = (k >= 0) ? k : (k ^ 0x7FFFFFFF);
    return __int_as_float(i);
}

__global__ void init_negmax_kernel(int* __restrict__ m, int n) {
    int t = blockIdx.x * blockDim.x + threadIdx.x;
    if (t < n) m[t] = f32_order(-INFINITY);
}

// alpha[e,h] = q[dst]·(k[src]+eproj) / sqrt(C); segment max via atomicMax
__global__ void attn_logit_kernel(const float* __restrict__ Q,
                                  const float* __restrict__ Kf,
                                  const float* __restrict__ Ep,
                                  const int* __restrict__ src,
                                  const int* __restrict__ dst,
                                  float* __restrict__ alpha,
                                  int* __restrict__ m,
                                  int E, int H, int C)
{
    int t = blockIdx.x * blockDim.x + threadIdx.x;
    if (t >= E * H) return;
    int ei = t / H, h = t - ei * H;
    int s = src[ei], d = dst[ei];
    const float* qp = Q  + ((size_t)d * H + h) * C;
    const float* kp = Kf + ((size_t)s * H + h) * C;
    const float* ep = Ep + ((size_t)ei * H + h) * C;
    float acc = 0.0f;
    for (int c = 0; c < C; c += 4) {     // C is 32 or 64, rows 16B-aligned
        v4f qv = *(const v4f*)(qp + c);
        v4f kv = *(const v4f*)(kp + c);
        v4f ev = *(const v4f*)(ep + c);
        acc += qv[0] * (kv[0] + ev[0]) + qv[1] * (kv[1] + ev[1])
             + qv[2] * (kv[2] + ev[2]) + qv[3] * (kv[3] + ev[3]);
    }
    acc *= rsqrtf((float)C);
    alpha[t] = acc;
    atomicMax(&m[(size_t)d * H + h], f32_order(acc));
}

__global__ void attn_exp_kernel(float* __restrict__ alpha,
                                const int* __restrict__ dst,
                                const int* __restrict__ m,
                                float* __restrict__ den,
                                int E, int H)
{
    int t = blockIdx.x * blockDim.x + threadIdx.x;
    if (t >= E * H) return;
    int ei = t / H, h = t - ei * H;
    int d = dst[ei];
    float mv = order_f32(m[(size_t)d * H + h]);
    if (!isfinite(mv)) mv = 0.0f;
    float w = expf(alpha[t] - mv);
    alpha[t] = w;
    atomicAdd(&den[(size_t)d * H + h], w);
}

// agg[dst, h, c] += (v[src,h,c] + eproj[e,h,c]) * softmax_w
__global__ void attn_scatter_kernel(const float* __restrict__ V,
                                    const float* __restrict__ Ep,
                                    const float* __restrict__ alpha,
                                    const float* __restrict__ den,
                                    const int* __restrict__ src,
                                    const int* __restrict__ dst,
                                    float* __restrict__ agg,
                                    int E, int H, int C)
{
    int CT = H * C;
    int t = blockIdx.x * blockDim.x + threadIdx.x;
    if (t >= E * CT) return;
    int ei = t / CT, r = t - ei * CT;
    int h = r / C;
    int s = src[ei], d = dst[ei];
    float w = alpha[(size_t)ei * H + h] / (den[(size_t)d * H + h] + 1e-16f);
    float val = (V[(size_t)s * CT + r] + Ep[(size_t)ei * CT + r]) * w;
    atomicAdd(&agg[(size_t)d * CT + r], val);
}

// out = relu(mean_h(agg) + skip)
__global__ void combine_kernel(const float* __restrict__ agg,
                               const float* __restrict__ skip,
                               float* __restrict__ out,
                               int Nn, int H, int C)
{
    int t = blockIdx.x * blockDim.x + threadIdx.x;
    if (t >= Nn * C) return;
    int n = t / C, c = t - n * C;
    float s = 0.0f;
    for (int h = 0; h < H; ++h) s += agg[((size_t)n * H + h) * C + c];
    out[t] = fmaxf(s / (float)H + skip[(size_t)n * C + c], 0.0f);
}

// ============================ pooling ================================
__global__ void pool_sum_kernel(const float* __restrict__ h,
                                const int* __restrict__ batch,
                                float* __restrict__ g, int Nn, int C)
{
    int t = blockIdx.x * blockDim.x + threadIdx.x;
    if (t >= Nn * C) return;
    int n = t / C, c = t - n * C;
    atomicAdd(&g[(size_t)batch[n] * C + c], h[t]);
}
__global__ void pool_cnt_kernel(const int* __restrict__ batch,
                                float* __restrict__ cnt, int Nn)
{
    int t = blockIdx.x * blockDim.x + threadIdx.x;
    if (t < Nn) atomicAdd(&cnt[batch[t]], 1.0f);
}
__global__ void pool_div_kernel(float* __restrict__ g,
                                const float* __restrict__ cnt, int Ng, int C)
{
    int t = blockIdx.x * blockDim.x + threadIdx.x;
    if (t >= Ng * C) return;
    g[t] /= fmaxf(cnt[t / C], 1.0f);
}

// =========================== host side ===============================
static inline void launch_gemm(const float* A, const float* W, const float* b,
                               float* C, int M, int K, int N, int relu,
                               hipStream_t st)
{
    dim3 grid((unsigned)((M + 63) / 64), (unsigned)((N + 63) / 64));
    gemm_bf16_wmma<<<grid, dim3(128), 0, st>>>(A, W, b, C, M, K, N, relu);
}

struct ConvScratch {
    float *Q, *K, *V, *Ep, *Skip, *Agg, *Alpha, *Den;
    int   *M;
};

static void run_conv(const float* xin, int Cin,
                     const float* qW, const float* qb,
                     const float* kW, const float* kb,
                     const float* vW, const float* vb,
                     const float* eW, int Fe,
                     const float* skW, const float* skb,
                     const float* eattr, const int* src, const int* dst,
                     int H, int C, const ConvScratch& S, float* out,
                     hipStream_t st)
{
    int CT = H * C;
    // dense projections (WMMA)
    launch_gemm(xin,   qW, qb,      S.Q,    NN, Cin, CT, 0, st);
    launch_gemm(xin,   kW, kb,      S.K,    NN, Cin, CT, 0, st);
    launch_gemm(xin,   vW, vb,      S.V,    NN, Cin, CT, 0, st);
    launch_gemm(eattr, eW, nullptr, S.Ep,   NE, Fe,  CT, 0, st);
    launch_gemm(xin,   skW, skb,    S.Skip, NN, Cin, C,  0, st);

    hipMemsetAsync(S.Agg, 0, (size_t)NN * CT * sizeof(float), st);
    hipMemsetAsync(S.Den, 0, (size_t)NN * H * sizeof(float), st);
    init_negmax_kernel<<<(NN * H + 255) / 256, 256, 0, st>>>(S.M, NN * H);

    int eh = NE * H;
    attn_logit_kernel<<<(eh + 255) / 256, 256, 0, st>>>(
        S.Q, S.K, S.Ep, src, dst, S.Alpha, S.M, NE, H, C);
    attn_exp_kernel<<<(eh + 255) / 256, 256, 0, st>>>(
        S.Alpha, dst, S.M, S.Den, NE, H);
    int ehc = NE * CT;
    attn_scatter_kernel<<<(ehc + 255) / 256, 256, 0, st>>>(
        S.V, S.Ep, S.Alpha, S.Den, src, dst, S.Agg, NE, H, C);
    int nc = NN * C;
    combine_kernel<<<(nc + 255) / 256, 256, 0, st>>>(S.Agg, S.Skip, out, NN, H, C);
}

extern "C" void kernel_launch(void* const* d_in, const int* in_sizes, int n_in,
                              void* d_out, int out_size, void* d_ws, size_t ws_size,
                              hipStream_t stream)
{
    (void)in_sizes; (void)n_in; (void)out_size; (void)ws_size;

    const float* x     = (const float*)d_in[0];
    const int*   eidx  = (const int*)d_in[1];
    const float* eattr = (const float*)d_in[2];
    const int*   batch = (const int*)d_in[3];
    const int* src = eidx;        // edge_index[0]
    const int* dst = eidx + NE;   // edge_index[1]

    // params: dict keys flattened alphabetically (jax pytree order)
    const float* c1_eW  = (const float*)d_in[4];
    const float* c1_kW  = (const float*)d_in[5];
    const float* c1_kb  = (const float*)d_in[6];
    const float* c1_qW  = (const float*)d_in[7];
    const float* c1_qb  = (const float*)d_in[8];
    const float* c1_skW = (const float*)d_in[9];
    const float* c1_skb = (const float*)d_in[10];
    const float* c1_vW  = (const float*)d_in[11];
    const float* c1_vb  = (const float*)d_in[12];
    const float* c2_eW  = (const float*)d_in[13];
    const float* c2_kW  = (const float*)d_in[14];
    const float* c2_kb  = (const float*)d_in[15];
    const float* c2_qW  = (const float*)d_in[16];
    const float* c2_qb  = (const float*)d_in[17];
    const float* c2_skW = (const float*)d_in[18];
    const float* c2_skb = (const float*)d_in[19];
    const float* c2_vW  = (const float*)d_in[20];
    const float* c2_vb  = (const float*)d_in[21];
    const float* fc1W   = (const float*)d_in[22];
    const float* fc1b   = (const float*)d_in[23];
    const float* fc2W   = (const float*)d_in[24];
    const float* fc2b   = (const float*)d_in[25];
    const float* fc3W   = (const float*)d_in[26];
    const float* fc3b   = (const float*)d_in[27];
    const float* fc4W   = (const float*)d_in[28];
    const float* fc4b   = (const float*)d_in[29];

    // ---- workspace bump allocator (sized for conv2: H=4, C=64, CT=256) ----
    char* ws = (char*)d_ws;
    size_t off = 0;
    auto alloc = [&](size_t bytes) -> void* {
        void* p = ws + off;
        off += (bytes + 255) & ~(size_t)255;
        return p;
    };
    ConvScratch S;
    S.Q     = (float*)alloc((size_t)NN * 256 * 4);
    S.K     = (float*)alloc((size_t)NN * 256 * 4);
    S.V     = (float*)alloc((size_t)NN * 256 * 4);
    S.Ep    = (float*)alloc((size_t)NE * 256 * 4);
    S.Skip  = (float*)alloc((size_t)NN * 64 * 4);
    S.Agg   = (float*)alloc((size_t)NN * 256 * 4);
    S.Alpha = (float*)alloc((size_t)NE * 4 * 4);
    S.Den   = (float*)alloc((size_t)NN * 4 * 4);
    S.M     = (int*)  alloc((size_t)NN * 4 * 4);
    float* H1  = (float*)alloc((size_t)NN * 32 * 4);
    float* H2  = (float*)alloc((size_t)NN * 64 * 4);
    float* G   = (float*)alloc((size_t)NG * 64 * 4);
    float* Cnt = (float*)alloc((size_t)NG * 4);
    float* F1  = (float*)alloc((size_t)NG * 1024 * 4);
    float* F2  = (float*)alloc((size_t)NG * 1024 * 4);

    // ---- conv1: 64 -> HID=32, 4 heads (CT=128) ----
    run_conv(x, 64, c1_qW, c1_qb, c1_kW, c1_kb, c1_vW, c1_vb,
             c1_eW, 16, c1_skW, c1_skb, eattr, src, dst,
             4, 32, S, H1, stream);

    // ---- conv2: 32 -> 64, 4 heads (CT=256) ----
    run_conv(H1, 32, c2_qW, c2_qb, c2_kW, c2_kb, c2_vW, c2_vb,
             c2_eW, 16, c2_skW, c2_skb, eattr, src, dst,
             4, 64, S, H2, stream);

    // ---- global mean pool ----
    hipMemsetAsync(G,   0, (size_t)NG * 64 * sizeof(float), stream);
    hipMemsetAsync(Cnt, 0, (size_t)NG * sizeof(float), stream);
    pool_sum_kernel<<<(NN * 64 + 255) / 256, 256, 0, stream>>>(H2, batch, G, NN, 64);
    pool_cnt_kernel<<<(NN + 255) / 256, 256, 0, stream>>>(batch, Cnt, NN);
    pool_div_kernel<<<(NG * 64 + 255) / 256, 256, 0, stream>>>(G, Cnt, NG, 64);

    // ---- MLP head (WMMA) ----
    launch_gemm(G,  fc1W, fc1b, F1, NG, 64,   1024, 1, stream);
    launch_gemm(F1, fc2W, fc2b, F2, NG, 1024, 1024, 1, stream);
    launch_gemm(F2, fc3W, fc3b, F1, NG, 1024, 512,  1, stream);
    launch_gemm(F1, fc4W, fc4b, (float*)d_out, NG, 512, 3, 0, stream);
}
